// BindingAffinityGAT_35278861369834
// MI455X (gfx1250) — matrix operand order repaired and checked by reading, not compiled, not run on hardware
//
#include <hip/hip_runtime.h>
#include <hip/hip_bf16.h>

#define Hh 8
#define Cc 128
#define Dd 1024
#define NEGS 0.2f
#define EPSB 1e-5f

typedef __attribute__((ext_vector_type(16))) __bf16 v16bf;
typedef __attribute__((ext_vector_type(8)))  __bf16 v8bf;
typedef __attribute__((ext_vector_type(8)))  float  v8f;
typedef __attribute__((ext_vector_type(4)))  unsigned v4u;
typedef __attribute__((ext_vector_type(8)))  int   v8i;
typedef __attribute__((ext_vector_type(4)))  int   v4i;

#if __has_builtin(__builtin_amdgcn_tensor_load_to_lds)
#define HAVE_TDM 1
#else
#define HAVE_TDM 0
#endif

__device__ __forceinline__ __bf16 f2bf(float f) {
  union { unsigned short u; __bf16 b; } cv;
  unsigned x = __float_as_uint(f);
  cv.u = (unsigned short)((x + 0x7FFFu + ((x >> 16) & 1u)) >> 16);  // RNE
  return cv.b;
}
__device__ __forceinline__ unsigned fenc(float f) {
  unsigned u = __float_as_uint(f);
  return (u & 0x80000000u) ? ~u : (u | 0x80000000u);
}
__device__ __forceinline__ float fdec(unsigned u) {
  unsigned v = (u & 0x80000000u) ? (u & 0x7FFFFFFFu) : ~u;
  return __uint_as_float(v);
}

// ================= WMMA GEMM with TDM-staged weight panel =================
// C[M x Nn] = A[M x K](bf16) * Bt[Nn x K](bf16, pre-transposed) + bias
// M multiple of 512 (caller pads; operands live in padded workspace),
// K multiple of 32, Nn multiple of 64.
// Block = 256 thr (8 waves) covering 512 M-rows x 64 N-cols.
// Wave tile 64x64 = 4x4 accumulators, branch-free inner loop.
// B panel (64 cols x 32 K) staged in LDS by the Tensor Data Mover,
// double-buffered, padded 64B->80B rows (bank stride 20 ->
// conflict-free ds_load_b128 fragment reads).

#define BM 512
#define WM 64
#define BN 64
#define KC 32
#define LDSROW 80

#if HAVE_TDM
__device__ __forceinline__ void tdm_issue_b(const __bf16* Bt, int koff, int row0,
                                            int K, int Nn, unsigned ldsOff) {
  unsigned long long ga = (unsigned long long)(const void*)(Bt + (size_t)row0 * K + koff);
  v4u g0;
  g0[0] = 1u;                                              // count=1 (user D#)
  g0[1] = ldsOff;                                          // lds_addr (bytes)
  g0[2] = (unsigned)(ga & 0xFFFFFFFFu);                    // global_addr[31:0]
  g0[3] = (unsigned)((ga >> 32) & 0x01FFFFFFu) | (2u << 30); // addr[56:32] | type=2
  unsigned td0 = (unsigned)K, td1 = (unsigned)Nn;
  unsigned long long s0 = (unsigned long long)K;           // dim0 stride (elements)
  v8i g1;
  g1[0] = (int)((1u << 16)      // data_size = 2 bytes
              | (1u << 20)      // pad_enable
              | (3u << 22)      // pad_interval: every 16 DWORDs (one 64B row)
              | (3u << 25));    // pad_amount: 4 DWORDs (16B) -> 80B row stride
  g1[1] = (int)((td0 & 0xFFFFu) << 16);
  g1[2] = (int)((td0 >> 16) | ((td1 & 0xFFFFu) << 16));
  g1[3] = (int)((td1 >> 16) | ((unsigned)KC << 16));       // tile_dim0 = 32 elems
  g1[4] = (int)((unsigned)BN);                             // tile_dim1 = 64, tile_dim2 = 0
  g1[5] = (int)(s0 & 0xFFFFFFFFu);
  g1[6] = (int)((s0 >> 32) & 0xFFFFu);
  g1[7] = 0;
  v4i z4 = {0, 0, 0, 0};
  v8i z8 = {0, 0, 0, 0, 0, 0, 0, 0};
  // 6-arg toolchain variant: (g0, g1, g2, g3, extra, cpol)
  __builtin_amdgcn_tensor_load_to_lds(g0, g1, z4, z4, z8, 0);
}
#endif

__global__ __launch_bounds__(256) void k_wmma_gemm(
    const __bf16* __restrict__ A, const __bf16* __restrict__ Bt,
    const float* __restrict__ bias, float* __restrict__ C,
    int M, int K, int Nn, int act)
{
  __shared__ __align__(16) unsigned char Bs[2][BN * LDSROW];  // 2 x 5 KB

  const int wave = threadIdx.x >> 5;
  const int lane = threadIdx.x & 31;
  const int half = lane >> 4;
  const int l15  = lane & 15;

  const int mblocks = M / BM;                 // M padded to BM by caller
  const int bm = blockIdx.x % mblocks;
  const int bn = blockIdx.x / mblocks;
  const int tn0 = bn * BN;
  const int tm0 = bm * BM + wave * WM;
  const int nk  = K / KC;

  v8f zero8 = {0.f,0.f,0.f,0.f,0.f,0.f,0.f,0.f};
  v8f acc[4][4];
#pragma unroll
  for (int i = 0; i < 4; ++i)
#pragma unroll
    for (int j = 0; j < 4; ++j) acc[i][j] = zero8;

#if HAVE_TDM
  const unsigned lds0 = (unsigned)(unsigned long long)(void*)&Bs[0][0];
  const unsigned lds1 = (unsigned)(unsigned long long)(void*)&Bs[1][0];
  if (wave == 0) {                       // prologue: DMA chunk 0 into buf 0
    tdm_issue_b(Bt, 0, tn0, K, Nn, lds0);
    __builtin_amdgcn_s_wait_tensorcnt(0);
  }
  __syncthreads();
#endif

  for (int kc = 0; kc < nk; ++kc) {
#if HAVE_TDM
    const int buf = kc & 1;
    if (wave == 0 && (kc + 1) < nk)      // DMA next chunk into other buffer
      tdm_issue_b(Bt, (kc + 1) * KC, tn0, K, Nn, buf ? lds0 : lds1);
#else
    const int buf = 0;
    __syncthreads();
    {  // cooperative stage: 64 rows x 64 B, 16 B per thread
      int row = threadIdx.x >> 2, seg = threadIdx.x & 3;
      v8bf v = *(const v8bf*)(Bt + (size_t)(tn0 + row) * K + kc * KC + seg * 8);
      *(v8bf*)(&Bs[0][row * LDSROW + seg * 16]) = v;
    }
    __syncthreads();
#endif

    // A fragments: row M = lane%16; K contiguous: k = 16*half + s
    v16bf afrag[4];
#pragma unroll
    for (int i = 0; i < 4; ++i) {
      const __bf16* arow = A + (size_t)(tm0 + i * 16 + l15) * K + kc * KC;
      v8bf alo = *(const v8bf*)(arow + half * 16);
      v8bf ahi = *(const v8bf*)(arow + half * 16 + 8);
#pragma unroll
      for (int e = 0; e < 8; ++e) { afrag[i][e] = alo[e]; afrag[i][e + 8] = ahi[e]; }
    }

    const unsigned char* bbase = Bs[buf];
#pragma unroll
    for (int j = 0; j < 4; ++j) {
      v16bf b = *(const v16bf*)(bbase + (j * 16 + l15) * LDSROW + half * 32);
#pragma unroll
      for (int i = 0; i < 4; ++i)
        acc[i][j] = __builtin_amdgcn_wmma_f32_16x16x32_bf16(
            false, afrag[i], false, b, (short)0, acc[i][j], false, false);
    }

#if HAVE_TDM
    if (wave == 0) __builtin_amdgcn_s_wait_tensorcnt(0);
    __syncthreads();
#endif
  }

#pragma unroll
  for (int i = 0; i < 4; ++i) {
#pragma unroll
    for (int j = 0; j < 4; ++j) {
      int col = tn0 + j * 16 + l15;
      float bv = bias ? bias[col] : 0.0f;
#pragma unroll
      for (int r = 0; r < 8; ++r) {
        int row = tm0 + i * 16 + half * 8 + r;
        float v = acc[i][j][r] + bv;
        if (act == 1) v = fmaxf(v, 0.0f);
        C[(size_t)row * Nn + col] = v;
      }
    }
  }
}

// ---------------- helpers ----------------
__global__ void k_fill_u32(unsigned* p, unsigned v, long n) {
  long i = (long)blockIdx.x * blockDim.x + threadIdx.x;
  if (i < n) p[i] = v;
}

__global__ void k_f32_to_bf16_pad(const float* __restrict__ s, __bf16* __restrict__ d,
                                  int rows, int K, int Kpad) {
  long i = (long)blockIdx.x * blockDim.x + threadIdx.x;
  if (i >= (long)rows * Kpad) return;
  int r = (int)(i / Kpad), k = (int)(i % Kpad);
  d[i] = (k < K) ? f2bf(s[(size_t)r * K + k]) : f2bf(0.0f);
}

// W[K x Nn] (row major) -> Wt[Nn x Kpad] bf16, zero-padded K
__global__ void k_w_to_bf16_t(const float* __restrict__ W, __bf16* __restrict__ Wt,
                              int K, int Kpad, int Nn) {
  long i = (long)blockIdx.x * blockDim.x + threadIdx.x;
  if (i >= (long)Nn * Kpad) return;
  int n = (int)(i / Kpad), k = (int)(i % Kpad);
  Wt[i] = (k < K) ? f2bf(W[(size_t)k * Nn + n]) : f2bf(0.0f);
}

__global__ void k_build_edges(const int* __restrict__ ei, int* __restrict__ src,
                              int* __restrict__ dst, int E, int Etot) {
  int i = blockIdx.x * blockDim.x + threadIdx.x;
  if (i >= Etot) return;
  if (i < E) { src[i] = ei[i]; dst[i] = ei[E + i]; }
  else       { src[i] = i - E; dst[i] = i - E; }   // self loops
}

// ---------------- edge phase ----------------
__global__ void k_edge_scores(const float* __restrict__ xl, const float* __restrict__ xr,
                              const int* __restrict__ src, const int* __restrict__ dst,
                              const float* __restrict__ att, float* __restrict__ e, int Etot) {
  long i = (long)blockIdx.x * blockDim.x + threadIdx.x;
  if (i >= (long)Etot * Hh) return;
  int eid = (int)(i >> 3), h = (int)(i & 7);
  const float4* pl = (const float4*)(xl + (size_t)src[eid] * Dd + h * Cc);
  const float4* pr = (const float4*)(xr + (size_t)dst[eid] * Dd + h * Cc);
  const float4* pa = (const float4*)(att + h * Cc);
  float s = 0.0f;
#pragma unroll 4
  for (int c = 0; c < Cc / 4; ++c) {
    float4 a = pl[c], b = pr[c], w = pa[c];
    float z0 = a.x + b.x, z1 = a.y + b.y, z2 = a.z + b.z, z3 = a.w + b.w;
    z0 = (z0 > 0.f) ? z0 : NEGS * z0;
    z1 = (z1 > 0.f) ? z1 : NEGS * z1;
    z2 = (z2 > 0.f) ? z2 : NEGS * z2;
    z3 = (z3 > 0.f) ? z3 : NEGS * z3;
    s += z0 * w.x + z1 * w.y + z2 * w.z + z3 * w.w;
  }
  e[i] = s;
}

__global__ void k_seg_max(const float* __restrict__ e, const int* __restrict__ dst,
                          unsigned* __restrict__ mu, int Etot) {
  long i = (long)blockIdx.x * blockDim.x + threadIdx.x;
  if (i >= (long)Etot * Hh) return;
  int eid = (int)(i >> 3), h = (int)(i & 7);
  atomicMax(&mu[(size_t)dst[eid] * Hh + h], fenc(e[i]));
}

__global__ void k_max_finalize(unsigned* __restrict__ mu, long n) {  // decode in place
  long i = (long)blockIdx.x * blockDim.x + threadIdx.x;
  if (i >= n) return;
  unsigned u = mu[i];
  float v = (u == 0u) ? 0.0f : fdec(u);
  if (!__builtin_isfinite(v)) v = 0.0f;
  ((float*)mu)[i] = v;
}

__global__ void k_edge_w(float* __restrict__ e, const int* __restrict__ dst,
                         const float* __restrict__ m, float* __restrict__ den, int Etot) {
  long i = (long)blockIdx.x * blockDim.x + threadIdx.x;
  if (i >= (long)Etot * Hh) return;
  int eid = (int)(i >> 3), h = (int)(i & 7);
  float w = __expf(e[i] - m[(size_t)dst[eid] * Hh + h]);
  e[i] = w;
  atomicAdd(&den[(size_t)dst[eid] * Hh + h], w);
}

__global__ void k_scatter(const float* __restrict__ xl, const float* __restrict__ w,
                          const float* __restrict__ den, const int* __restrict__ src,
                          const int* __restrict__ dst, float* __restrict__ out) {
  int eid = blockIdx.x;
  int s = src[eid], d = dst[eid];
  for (int ch = threadIdx.x; ch < Dd; ch += blockDim.x) {
    int h = ch >> 7;
    float a = w[(size_t)eid * Hh + h] / fmaxf(den[(size_t)d * Hh + h], 1e-16f);
    atomicAdd(&out[(size_t)d * Dd + ch], a * xl[(size_t)s * Dd + ch]);
  }
}

// ---------------- post: bias + BN + ELU ----------------
__global__ void k_post_concat(float* __restrict__ h, const float* __restrict__ bias,
                              const float* __restrict__ g, const float* __restrict__ b,
                              const float* __restrict__ mn, const float* __restrict__ vr,
                              long n) {
  long i = (long)blockIdx.x * blockDim.x + threadIdx.x;
  if (i >= n) return;
  int j = (int)(i & (Dd - 1));
  float v = h[i] + bias[j];
  v = g[j] * (v - mn[j]) * __frsqrt_rn(vr[j] + EPSB) + b[j];
  h[i] = (v > 0.0f) ? v : (__expf(v) - 1.0f);
}

__global__ void k_post_mean(const float* __restrict__ acc, const float* __restrict__ bias,
                            const float* __restrict__ g, const float* __restrict__ b,
                            const float* __restrict__ mn, const float* __restrict__ vr,
                            float* __restrict__ h3, int N) {
  long i = (long)blockIdx.x * blockDim.x + threadIdx.x;
  if (i >= (long)N * Cc) return;
  int n = (int)(i >> 7), c = (int)(i & (Cc - 1));
  float s = 0.0f;
#pragma unroll
  for (int h = 0; h < Hh; ++h) s += acc[(size_t)n * Dd + h * Cc + c];
  float v = s * 0.125f + bias[c];
  v = g[c] * (v - mn[c]) * __frsqrt_rn(vr[c] + EPSB) + b[c];
  h3[i] = (v > 0.0f) ? v : (__expf(v) - 1.0f);
}

// ---------------- pooling + head ----------------
__global__ void k_gate2(const float* __restrict__ g1, const float* __restrict__ w2,
                        const float* __restrict__ b2, float* __restrict__ gv, int N) {
  int n = blockIdx.x * blockDim.x + threadIdx.x;
  if (n >= N) return;
  float s = b2[0];
#pragma unroll 4
  for (int j = 0; j < 64; ++j) s += g1[(size_t)n * 64 + j] * w2[j];
  gv[n] = s;
}

__global__ void k_pool_max(const float* __restrict__ gv, const int* __restrict__ batch,
                           unsigned* __restrict__ gmu, int N) {
  int n = blockIdx.x * blockDim.x + threadIdx.x;
  if (n >= N) return;
  atomicMax(&gmu[batch[n]], fenc(gv[n]));
}

__global__ void k_pool_w(float* __restrict__ gv, const int* __restrict__ batch,
                         const float* __restrict__ gm, float* __restrict__ gden, int N) {
  int n = blockIdx.x * blockDim.x + threadIdx.x;
  if (n >= N) return;
  float w = __expf(gv[n] - gm[batch[n]]);
  gv[n] = w;
  atomicAdd(&gden[batch[n]], w);
}

__global__ void k_pool_sum(const float* __restrict__ gv, const float* __restrict__ gden,
                           const int* __restrict__ batch, const float* __restrict__ h3,
                           float* __restrict__ pooled, int N) {
  long i = (long)blockIdx.x * blockDim.x + threadIdx.x;
  if (i >= (long)N * Cc) return;
  int n = (int)(i >> 7), c = (int)(i & (Cc - 1));
  int b = batch[n];
  float a = gv[n] / fmaxf(gden[b], 1e-16f);
  atomicAdd(&pooled[(size_t)b * Cc + c], a * h3[i]);
}

__global__ void k_fc3(const float* __restrict__ o2, const float* __restrict__ w,
                      const float* __restrict__ b, float* __restrict__ out, int B) {
  int i = blockIdx.x * blockDim.x + threadIdx.x;
  if (i >= B) return;
  float s = b[0];
#pragma unroll
  for (int j = 0; j < 64; ++j) s += o2[(size_t)i * 64 + j] * w[j];
  out[i] = s;
}

// =======================================================================================
extern "C" void kernel_launch(void* const* d_in, const int* in_sizes, int n_in,
                              void* d_out, int out_size, void* d_ws, size_t ws_size,
                              hipStream_t stream) {
  const int N    = in_sizes[2];          // 10000
  const int E    = in_sizes[1] / 2;      // 160000
  const int Etot = E + N;
  const int Din  = in_sizes[0] / N;      // 49
  const int B    = 16;
  const int Npad = ((N + BM - 1) / BM) * BM;   // 10240: padded GEMM row count
  const int Bpad = BM;                         // 512: padded head GEMM rows

  const float* x     = (const float*)d_in[0];
  const int*   ei    = (const int*)d_in[1];
  const int*   batch = (const int*)d_in[2];

  auto F = [&](int i) { return (const float*)d_in[i]; };

  // ---- workspace carve-up (GEMM-facing buffers padded to Npad/Bpad rows) ----
  char* base = (char*)d_ws;
  size_t off = 0;
  auto alloc = [&](size_t bytes) -> void* {
    void* p = base + off;
    off = (off + bytes + 255) & ~(size_t)255;
    return p;
  };
  float*  P0   = (float*)alloc((size_t)N * Dd * 4);
  float*  P1   = (float*)alloc((size_t)N * Dd * 4);
  float*  XL   = (float*)alloc((size_t)Npad * Dd * 4);
  float*  XR   = (float*)alloc((size_t)Npad * Dd * 4);
  float*  H3   = (float*)alloc((size_t)N * Cc * 4);
  __bf16* ABF  = (__bf16*)alloc((size_t)Npad * Dd * 2);
  __bf16* WLB  = (__bf16*)alloc((size_t)Dd * Dd * 2);
  __bf16* WRB  = (__bf16*)alloc((size_t)Dd * Dd * 2);
  float*  EW   = (float*)alloc((size_t)Etot * Hh * 4);
  unsigned* MU = (unsigned*)alloc((size_t)N * Hh * 4);
  float*  DEN  = (float*)alloc((size_t)N * Hh * 4);
  int*    SRC  = (int*)alloc((size_t)Etot * 4);
  int*    DST  = (int*)alloc((size_t)Etot * 4);
  float*  G1   = (float*)alloc((size_t)Npad * 64 * 4);
  float*  GV   = (float*)alloc((size_t)N * 4);
  unsigned* GMU = (unsigned*)alloc(64);
  float*  GDEN = (float*)alloc(64);
  float*  POOL = (float*)alloc((size_t)B * Cc * 4);
  __bf16* PBF  = (__bf16*)alloc((size_t)Bpad * Cc * 2);
  float*  O1   = (float*)alloc((size_t)Bpad * 128 * 4);
  __bf16* O1B  = (__bf16*)alloc((size_t)Bpad * 128 * 2);
  float*  O2   = (float*)alloc((size_t)Bpad * 64 * 4);

  auto cdivl = [](long a, long b) { return (unsigned)((a + b - 1) / b); };

  auto gemm = [&](const __bf16* A, const __bf16* Bt, const float* bias, float* C,
                  int M, int K, int Nn, int act) {
    int Mp = ((M + BM - 1) / BM) * BM;       // branch-free kernel: M padded
    unsigned blocks = (unsigned)(Mp / BM) * (unsigned)(Nn / BN);
    k_wmma_gemm<<<blocks, 256, 0, stream>>>(A, Bt, bias, C, Mp, K, Nn, act);
  };
  auto fill = [&](void* p, unsigned v, long nwords) {
    k_fill_u32<<<cdivl(nwords, 256), 256, 0, stream>>>((unsigned*)p, v, nwords);
  };

  // edges with self loops
  k_build_edges<<<cdivl(Etot, 256), 256, 0, stream>>>(ei, SRC, DST, E, Etot);

  // ---- one GATv2 layer ----
  auto gat_layer = [&](const float* hin, int din, int kpad, int pbase, int bnbase,
                       float* accbuf, bool concat) {
    k_f32_to_bf16_pad<<<cdivl((long)N * kpad, 256), 256, 0, stream>>>(hin, ABF, N, din, kpad);
    k_w_to_bf16_t<<<cdivl((long)Dd * kpad, 256), 256, 0, stream>>>(F(pbase + 0), WLB, din, kpad, Dd);
    k_w_to_bf16_t<<<cdivl((long)Dd * kpad, 256), 256, 0, stream>>>(F(pbase + 2), WRB, din, kpad, Dd);
    gemm(ABF, WLB, F(pbase + 1), XL, N, kpad, Dd, 0);   // xl = h@Wl + bl
    gemm(ABF, WRB, F(pbase + 3), XR, N, kpad, Dd, 0);   // xr = h@Wr + br
    k_edge_scores<<<cdivl((long)Etot * Hh, 256), 256, 0, stream>>>(XL, XR, SRC, DST,
                                                                   F(pbase + 4), EW, Etot);
    fill(MU, 0u, (long)N * Hh);
    fill(DEN, 0u, (long)N * Hh);
    k_seg_max<<<cdivl((long)Etot * Hh, 256), 256, 0, stream>>>(EW, DST, MU, Etot);
    k_max_finalize<<<cdivl((long)N * Hh, 256), 256, 0, stream>>>(MU, (long)N * Hh);
    k_edge_w<<<cdivl((long)Etot * Hh, 256), 256, 0, stream>>>(EW, DST, (const float*)MU, DEN, Etot);
    fill(accbuf, 0u, (long)N * Dd);
    k_scatter<<<Etot, 256, 0, stream>>>(XL, EW, DEN, SRC, DST, accbuf);
    if (concat) {
      k_post_concat<<<cdivl((long)N * Dd, 256), 256, 0, stream>>>(
          accbuf, F(pbase + 5), F(bnbase), F(bnbase + 1), F(bnbase + 2), F(bnbase + 3),
          (long)N * Dd);
    } else {
      k_post_mean<<<cdivl((long)N * Cc, 256), 256, 0, stream>>>(
          accbuf, F(pbase + 5), F(bnbase), F(bnbase + 1), F(bnbase + 2), F(bnbase + 3), H3, N);
    }
  };

  gat_layer(x,  Din, 64,   3,  9, P1, true);   // layer 1: x -> P1
  gat_layer(P1, Dd,  Dd,  13, 19, P0, true);   // layer 2: P1 -> P0
  gat_layer(P0, Dd,  Dd,  23, 29, P1, false);  // layer 3: P0 -> H3

  // ---- gate MLP + attention pooling ----
  k_f32_to_bf16_pad<<<cdivl((long)N * Cc, 256), 256, 0, stream>>>(H3, ABF, N, Cc, Cc);
  k_w_to_bf16_t<<<cdivl((long)64 * Cc, 256), 256, 0, stream>>>(F(33), WLB, Cc, Cc, 64);
  gemm(ABF, WLB, F(34), G1, N, Cc, 64, 1);                       // relu(h@gate_w1+b1)
  k_gate2<<<cdivl(N, 256), 256, 0, stream>>>(G1, F(35), F(36), GV, N);
  fill(GMU, 0u, 16);
  fill(GDEN, 0u, 16);
  fill(POOL, 0u, (long)B * Cc);
  k_pool_max<<<cdivl(N, 256), 256, 0, stream>>>(GV, batch, GMU, N);
  k_max_finalize<<<1, 32, 0, stream>>>(GMU, 16);
  k_pool_w<<<cdivl(N, 256), 256, 0, stream>>>(GV, batch, (const float*)GMU, GDEN, N);
  k_pool_sum<<<cdivl((long)N * Cc, 256), 256, 0, stream>>>(GV, GDEN, batch, H3, POOL, N);

  // ---- MLP head ----
  k_f32_to_bf16_pad<<<cdivl((long)B * Cc, 256), 256, 0, stream>>>(POOL, PBF, B, Cc, Cc);
  k_w_to_bf16_t<<<cdivl((long)128 * Cc, 256), 256, 0, stream>>>(F(37), WLB, Cc, Cc, 128);
  gemm(PBF, WLB, F(38), O1, B, Cc, 128, 1);                      // fc1 + relu
  k_f32_to_bf16_pad<<<cdivl((long)B * 128, 256), 256, 0, stream>>>(O1, O1B, B, 128, 128);
  k_w_to_bf16_t<<<cdivl((long)64 * 128, 256), 256, 0, stream>>>(F(39), WLB, 128, 128, 64);
  gemm(O1B, WLB, F(40), O2, B, 128, 64, 1);                      // fc2 + relu
  k_fc3<<<1, 32, 0, stream>>>(O2, F(41), F(42), (float*)d_out, B);
}
